// PairSelfAttentionBlock_31155692765640
// MI455X (gfx1250) — matrix operand order repaired and checked by reading, not compiled, not run on hardware
//
#include <hip/hip_runtime.h>
#include <hip/hip_bf16.h>

typedef __bf16 bf16;
typedef __attribute__((ext_vector_type(16))) __bf16 v16bf;
typedef __attribute__((ext_vector_type(8)))  float  v8f;

#define N_ATOMS 64
#define D_IN    64
#define H       4
#define D       32
#define S       (N_ATOMS * N_ATOMS)   // 4096
#define HD      (H * D)               // 128
#define NCHUNK  (S / 64)              // 64 key-chunks of 64

// ---------------------------------------------------------------------------
// Pass 1: QKV projection.  x(S,64) @ W(64,128) + b  -> bf16 Q/K row-major
// [h][s][d], V transposed [h][d][s].  One wave per 16x16 output tile,
// two K-chunks of 32 (v_wmma_f32_16x16x32_bf16, fp32 accumulate).
// ---------------------------------------------------------------------------
__global__ __launch_bounds__(256) void qkv_kernel(
    const float* __restrict__ x,
    const float* __restrict__ Wq, const float* __restrict__ bq,
    const float* __restrict__ Wk, const float* __restrict__ bk,
    const float* __restrict__ Wv, const float* __restrict__ bv,
    bf16* __restrict__ Qo, bf16* __restrict__ Ko, bf16* __restrict__ Vt)
{
    const int wave = threadIdx.x >> 5;
    const int lane = threadIdx.x & 31;
    const int half = lane >> 4;
    const int ln   = lane & 15;

    const int gid   = blockIdx.x * 8 + wave;   // 0..6143
    const int mat   = gid >> 11;               // 0=Q 1=K 2=V
    const int rem   = gid & 2047;
    const int mtile = rem >> 3;                // 0..255 (rows of 16)
    const int ntile = rem & 7;                 // 0..7   (cols of 16)

    const float* W    = (mat == 0) ? Wq : (mat == 1) ? Wk : Wv;
    const float* bias = (mat == 0) ? bq : (mat == 1) ? bk : bv;

    const int arow = mtile * 16 + ln;          // A-matrix row (both halves)
    const int ncol = ntile * 16 + ln;          // B/C column

    v8f acc = {};
    #pragma unroll
    for (int kc = 0; kc < 2; ++kc) {
        // A (16x32 bf16): lanes 0-15 kbase=0, lanes 16-31 kbase=16;
        // elems 0..15 -> k = kbase+e
        const int kabase = kc * 32 + (half ? 16 : 0);
        v16bf a;
        #pragma unroll
        for (int e = 0; e < 8; ++e) a[e]     = (bf16)x[arow * D_IN + kabase + e];
        #pragma unroll
        for (int e = 0; e < 8; ++e) a[e + 8] = (bf16)x[arow * D_IN + kabase + 8 + e];

        // B (32x16 bf16): lane = column n; lanes 0-15 k=0..15, 16-31 k=16..31
        const int kbbase = kc * 32 + (half ? 16 : 0);
        v16bf b;
        #pragma unroll
        for (int e = 0; e < 16; ++e) b[e] = (bf16)W[(kbbase + e) * HD + ncol];

        acc = __builtin_amdgcn_wmma_f32_16x16x32_bf16(false, a, false, b,
                                                      (short)0, acc, false, false);
    }

    const float bb = bias[ncol];
    const int hh = ncol >> 5;   // head
    const int dd = ncol & 31;   // dim within head
    // C/D layout: lane = column; VGPR r -> row r (lanes 0-15) or r+8 (16-31)
    #pragma unroll
    for (int r = 0; r < 8; ++r) {
        const int orow = mtile * 16 + half * 8 + r;
        const float v  = acc[r] + bb;
        if (mat == 0)      Qo[(hh * S + orow) * D + dd] = (bf16)v;
        else if (mat == 1) Ko[(hh * S + orow) * D + dd] = (bf16)v;
        else               Vt[(hh * D + dd) * S + orow] = (bf16)v;
    }
}

// ---------------------------------------------------------------------------
// Stage one 64-key chunk of K (4KB contiguous) and transposed-V (32 x 128B)
// into LDS.  256 threads x one b128 each per matrix.  On gfx1250 this uses
// the async-copy engine (GLOBAL_LOAD_ASYNC_TO_LDS_B128, tracked by ASYNCcnt)
// via inline asm: vdst = LDS byte offset (low 32 bits of the generic shared
// pointer), vaddr = 64-bit global address, GV mode.
// ---------------------------------------------------------------------------
__device__ __forceinline__ void stage_chunk(
    const bf16* __restrict__ Kg, const bf16* __restrict__ Vtg,
    int h, int base, int tid, bf16* kdstBase, bf16* vdstBase)
{
    const bf16* ksrc = Kg + (size_t)(h * S + base) * D + tid * 8;   // 16B each
    bf16*       kdst = kdstBase + tid * 8;
    const int   seg  = tid >> 3;            // V row (d = 0..31)
    const int   off  = (tid & 7) * 8;       // 8 threads x 16B = 128B row
    const bf16* vsrc = Vtg + (size_t)(h * D + seg) * S + base + off;
    bf16*       vdst = vdstBase + seg * 64 + off;
#if defined(__gfx1250__)
    const unsigned klds = (unsigned)(size_t)kdst;   // LDS byte offset
    const unsigned vlds = (unsigned)(size_t)vdst;
    asm volatile("global_load_async_to_lds_b128 %0, %1, off"
                 :: "v"(klds), "v"(ksrc) : "memory");
    asm volatile("global_load_async_to_lds_b128 %0, %1, off"
                 :: "v"(vlds), "v"(vsrc) : "memory");
#else
    *(int4*)kdst = *(const int4*)ksrc;
    *(int4*)vdst = *(const int4*)vsrc;
#endif
}

__device__ __forceinline__ void wait_async_keep2()
{
#if defined(__gfx1250__)
    asm volatile("s_wait_asynccnt 0x2" ::: "memory");
#endif
}

__device__ __forceinline__ void wait_async_all()
{
#if defined(__gfx1250__)
    asm volatile("s_wait_asynccnt 0x0" ::: "memory");
#endif
}

// ---------------------------------------------------------------------------
// Pass 2: flash attention.  8 waves/block, one 16-query tile per wave,
// online softmax over 64-key chunks.  K/V chunks double-buffered in LDS via
// async copies; per chunk: 4 WMMAs for scores, LDS transpose of P, 4 WMMAs
// for P@V.
// ---------------------------------------------------------------------------
__global__ __launch_bounds__(256) void attn_kernel(
    const bf16* __restrict__ Q, const bf16* __restrict__ K,
    const bf16* __restrict__ Vt, const int* __restrict__ amask,
    float* __restrict__ O)
{
    __shared__ int  smask[N_ATOMS];
    __shared__ bf16 kbuf[2][64 * D];      // 4KB per buffer: [key][d]
    __shared__ bf16 vbuf[2][D * 64];      // 4KB per buffer: [d][key]
    __shared__ bf16 pstage[8][16 * 64];   // per-wave P staging (2KB each)

    const int tid = threadIdx.x;
    if (tid < N_ATOMS) smask[tid] = amask[tid];

    const int wave = tid >> 5;
    const int lane = tid & 31;
    const int half = lane >> 4;
    const int ln   = lane & 15;

    const int h     = blockIdx.x >> 5;          // head
    const int qb    = blockIdx.x & 31;
    const int qbase = qb * 128 + wave * 16;

    // Load Q tile into A layout (row = ln, kbase = half*16, contiguous)
    const int   kaq    = half ? 16 : 0;
    const bf16* qrow_p = Q + (size_t)(h * S + qbase + ln) * D;
    v16bf aQ;
    #pragma unroll
    for (int e = 0; e < 8; ++e) aQ[e]     = qrow_p[kaq + e];
    #pragma unroll
    for (int e = 0; e < 8; ++e) aQ[e + 8] = qrow_p[kaq + 8 + e];

    float mrow[8], lrow[8];
    #pragma unroll
    for (int r = 0; r < 8; ++r) { mrow[r] = -INFINITY; lrow[r] = 0.0f; }
    v8f o0 = {}, o1 = {};

    const float scale = 0.17677669529663687f;   // 1/sqrt(32)
    const int   kb    = half ? 16 : 0;          // B-matrix k offset
    bf16*       myp   = &pstage[wave][0];

    // Prologue: stage chunk 0 into buffer 0
    stage_chunk(K, Vt, h, 0, tid, kbuf[0], vbuf[0]);

    for (int chunk = 0; chunk < NCHUNK; ++chunk) {
        const int base = chunk * 64;
        const int cur  = chunk & 1;

        // Issue next chunk into the other buffer, then wait for current.
        if (chunk + 1 < NCHUNK) {
            stage_chunk(K, Vt, h, base + 64, tid, kbuf[cur ^ 1], vbuf[cur ^ 1]);
            wait_async_keep2();   // only the newest 2 ops still in flight
        } else {
            wait_async_all();
        }
        __syncthreads();   // current buffer visible to all waves

        const bf16* klds = kbuf[cur];
        const bf16* vlds = vbuf[cur];

        // ---- scores: 4 tiles of 16 keys, S = Q(16x32) @ K^T(32x16) ----
        v8f s[4];
        #pragma unroll
        for (int t = 0; t < 4; ++t) {
            const int   krow = t * 16 + ln;           // B column = key
            const bf16* kp   = klds + krow * D + kb;
            v16bf bK;
            #pragma unroll
            for (int e = 0; e < 16; ++e) bK[e] = kp[e];
            v8f z = {};
            s[t] = __builtin_amdgcn_wmma_f32_16x16x32_bf16(false, aQ, false, bK,
                                                           (short)0, z, false, false);
            const int key = base + krow;
            const int pv  = smask[key >> 6] & smask[key & 63];
            #pragma unroll
            for (int r = 0; r < 8; ++r)
                s[t][r] = pv ? s[t][r] * scale : -1.0e9f;
        }

        // ---- chunk row-max (lanes hold columns; reduce across 16-lane half)
        float cm[8];
        #pragma unroll
        for (int r = 0; r < 8; ++r) {
            float m = fmaxf(fmaxf(s[0][r], s[1][r]), fmaxf(s[2][r], s[3][r]));
            m = fmaxf(m, __shfl_xor(m, 1, 32));
            m = fmaxf(m, __shfl_xor(m, 2, 32));
            m = fmaxf(m, __shfl_xor(m, 4, 32));
            m = fmaxf(m, __shfl_xor(m, 8, 32));
            cm[r] = m;
        }

        // ---- online rescale ----
        #pragma unroll
        for (int r = 0; r < 8; ++r) {
            const float mnew = fmaxf(mrow[r], cm[r]);
            const float corr = __expf(mrow[r] - mnew);
            mrow[r] = mnew;
            lrow[r] *= corr;
            o0[r]   *= corr;
            o1[r]   *= corr;
        }

        // ---- probabilities -> LDS (C-layout scatter), row sums ----
        float rs[8];
        #pragma unroll
        for (int r = 0; r < 8; ++r) rs[r] = 0.0f;
        #pragma unroll
        for (int t = 0; t < 4; ++t) {
            #pragma unroll
            for (int r = 0; r < 8; ++r) {
                const float p = __expf(s[t][r] - mrow[r]);
                rs[r] += p;
                myp[(half * 8 + r) * 64 + t * 16 + ln] = (bf16)p;
            }
        }
        #pragma unroll
        for (int r = 0; r < 8; ++r) {
            float v = rs[r];
            v += __shfl_xor(v, 1, 32);
            v += __shfl_xor(v, 2, 32);
            v += __shfl_xor(v, 4, 32);
            v += __shfl_xor(v, 8, 32);
            lrow[r] += v;
        }

        // ---- O += P(16x64) @ V(64x32): 2 key sub-chunks x 2 d-tiles ----
        #pragma unroll
        for (int a = 0; a < 2; ++a) {
            const int   kap = half ? 16 : 0;
            const bf16* pr  = myp + ln * 64 + a * 32 + kap;   // A-layout read
            v16bf aP;
            #pragma unroll
            for (int e = 0; e < 8; ++e) aP[e]     = pr[e];
            #pragma unroll
            for (int e = 0; e < 8; ++e) aP[e + 8] = pr[8 + e];

            const int keyoff = a * 32 + kb;
            {
                const bf16* vp = vlds + ln * 64 + keyoff;          // d = ln
                v16bf bV;
                #pragma unroll
                for (int e = 0; e < 16; ++e) bV[e] = vp[e];
                o0 = __builtin_amdgcn_wmma_f32_16x16x32_bf16(false, aP, false, bV,
                                                             (short)0, o0, false, false);
            }
            {
                const bf16* vp = vlds + (16 + ln) * 64 + keyoff;   // d = 16+ln
                v16bf bV;
                #pragma unroll
                for (int e = 0; e < 16; ++e) bV[e] = vp[e];
                o1 = __builtin_amdgcn_wmma_f32_16x16x32_bf16(false, aP, false, bV,
                                                             (short)0, o1, false, false);
            }
        }

        __syncthreads();   // all reads done before buffer is re-staged
    }

    // ---- normalize and store O[h][s][d] fp32 ----
    #pragma unroll
    for (int r = 0; r < 8; ++r) {
        const float inv = 1.0f / lrow[r];
        const int   row = qbase + half * 8 + r;
        O[(size_t)(h * S + row) * D + ln]      = o0[r] * inv;
        O[(size_t)(h * S + row) * D + 16 + ln] = o1[r] * inv;
    }
}

// ---------------------------------------------------------------------------
// Pass 3: out[i, h*32+d] = sum_j O[h][i*64+j][d]
// ---------------------------------------------------------------------------
__global__ __launch_bounds__(128) void reduce_kernel(
    const float* __restrict__ O, float* __restrict__ out)
{
    const int idx = blockIdx.x * blockDim.x + threadIdx.x;  // 0..8191
    if (idx >= N_ATOMS * HD) return;
    const int i = idx / HD;
    const int c = idx % HD;
    const int h = c >> 5;
    const int d = c & 31;
    float acc = 0.0f;
    const float* p = O + (size_t)(h * S + i * 64) * D + d;
    #pragma unroll 8
    for (int j = 0; j < 64; ++j) acc += p[j * D];
    out[idx] = acc;
}

extern "C" void kernel_launch(void* const* d_in, const int* in_sizes, int n_in,
                              void* d_out, int out_size, void* d_ws, size_t ws_size,
                              hipStream_t stream)
{
    const float* pair  = (const float*)d_in[0];
    const int*   amask = (const int*)d_in[1];
    const float* Wq    = (const float*)d_in[2];
    const float* bq    = (const float*)d_in[3];
    const float* Wk    = (const float*)d_in[4];
    const float* bk    = (const float*)d_in[5];
    const float* Wv    = (const float*)d_in[6];
    const float* bv    = (const float*)d_in[7];

    char* ws = (char*)d_ws;
    bf16*  Q  = (bf16*)(ws);                         // 1 MB  (4*4096*32 bf16)
    bf16*  K  = (bf16*)(ws + (size_t)(1u << 20));    // 1 MB
    bf16*  Vt = (bf16*)(ws + (size_t)(2u << 20));    // 1 MB (transposed V)
    float* O  = (float*)(ws + (size_t)(3u << 20));   // 2 MB  (4*4096*32 fp32)

    // Pass 1: 3 matrices * 256 m-tiles * 8 n-tiles = 6144 waves / 8 per block
    qkv_kernel<<<768, 256, 0, stream>>>(pair, Wq, bq, Wk, bk, Wv, bv, Q, K, Vt);
    // Pass 2: 4 heads * 32 query-blocks of 128 rows
    attn_kernel<<<128, 256, 0, stream>>>(Q, K, Vt, amask, O);
    // Pass 3: 64*128 outputs
    reduce_kernel<<<64, 128, 0, stream>>>(O, (float*)d_out);
}
